// WindowMultiHeadAttention_39633958207866
// MI455X (gfx1250) — matrix operand order repaired and checked by reading, not compiled, not run on hardware
//
#include <hip/hip_runtime.h>

// ---------------------------------------------------------------------------
// Window MHA (faithful to reference incl. the transpose-reshape shuffle).
// B=4096 windows, N=64 tokens, C=256 channels. f16 WMMA, f32 accumulate.
// ---------------------------------------------------------------------------

#define N_TOK   64
#define D_MODEL 256
#define N_WIN   4096
#define ATTN_SCALE 0.17677669529663687f   // 32^-0.5

typedef __attribute__((ext_vector_type(16))) _Float16 v16h;
typedef __attribute__((ext_vector_type(8)))  _Float16 v8h;
typedef __attribute__((ext_vector_type(8)))  float    v8f;
typedef __attribute__((ext_vector_type(4)))  float    v4f;

// ---- LDS geometry (byte offsets into one shared block) --------------------
// region0 (33792 B) is time-multiplexed: xstage(f16 64x264) -> scores(f32
// 64x65) -> feats2(f16 64x264). All strides keep 16B alignment of fragments.
#define SX_STRIDE  264   // f16 elems per row (256 + 8 pad)
#define SVT_STRIDE 72    // f16 elems per row (64 + 8 pad)
#define SS_STRIDE  65    // f32 elems per row (65 mod 64 == 1 -> conflict free)
#define SA_STRIDE  72    // f16 elems per row
#define OFF_SQ     33792
#define OFF_SK     67584
#define OFF_SVT    101376
#define OFF_SA     138240
#define SMEM_BYTES 147456

// ---------------------------------------------------------------------------
// WMMA helpers (wave32 layouts per CDNA5 ISA 7.12.2)
// ---------------------------------------------------------------------------
__device__ __forceinline__ v16h cat8(v8h lo, v8h hi) {
  return __builtin_shufflevector(lo, hi, 0,1,2,3,4,5,6,7,8,9,10,11,12,13,14,15);
}

__device__ __forceinline__ v8f wmma_f16(v16h a, v16h b, v8f c) {
  // (neg_a, A, neg_b, B, c_mod, C, reuse_a, reuse_b)
  return __builtin_amdgcn_wmma_f32_16x16x32_f16(false, a, false, b, (short)0, c,
                                                false, false);
}

// A fragment: 16x32 f16 from row-major src (K contiguous).
// Lane group g: VGPR0-3 hold K = g*8 + 0..7; VGPR4-7 hold K = 16 + g*8 + 0..7.
__device__ __forceinline__ v16h load_a_frag(const _Float16* src, int stride,
                                            int row0, int kbase, int lane) {
  int m = lane & 15, g = lane >> 4;
  const _Float16* p = src + (size_t)(row0 + m) * stride + kbase + g * 8;
  v8h lo = *(const v8h*)(p);
  v8h hi = *(const v8h*)(p + 16);
  return cat8(lo, hi);
}

// B fragment: 32x16 f16 where B[k][n] = src[(col0+n)*stride + kbase + k]
// (i.e. src row-major with K contiguous within a row).
// Lane group g, col n: elements 0-7 are K = g*8 + 0..7; 8-15 are K = 16 + g*8 + 0..7.
__device__ __forceinline__ v16h load_b_frag(const _Float16* src, int stride,
                                            int col0, int kbase, int lane) {
  int n = lane & 15, g = lane >> 4;
  const _Float16* p = src + (size_t)(col0 + n) * stride + kbase + g * 8;
  v8h lo = *(const v8h*)(p);
  v8h hi = *(const v8h*)(p + 16);
  return cat8(lo, hi);
}

// ---------------------------------------------------------------------------
// Weight pre-conversion: fp32 -> f16, once per launch (weights then live in
// the 192MB L2 and are streamed as B-fragments by every block).
// ---------------------------------------------------------------------------
__global__ void convert_weights_kernel(const float* __restrict__ wq,
                                       const float* __restrict__ wk,
                                       const float* __restrict__ wv,
                                       const float* __restrict__ wp,
                                       _Float16* __restrict__ oq,
                                       _Float16* __restrict__ ok,
                                       _Float16* __restrict__ ov,
                                       _Float16* __restrict__ op) {
  int i = blockIdx.x * blockDim.x + threadIdx.x;   // 65536 threads
  oq[i] = (_Float16)wq[i];
  ok[i] = (_Float16)wk[i];
  ov[i] = (_Float16)wv[i];
  op[i] = (_Float16)wp[i];
}

// ---------------------------------------------------------------------------
// One projection pass: y = x @ W.T + b, result to LDS (mode-specialized so
// every global pointer is concrete -> global_load_* instead of flat_load_*).
// MODE 0: qh (scaled), MODE 1: kh, MODE 2: vh stored transposed.
// ---------------------------------------------------------------------------
template <int MODE>
__device__ __forceinline__ void projection_pass(
    const float* __restrict__ x, const _Float16* __restrict__ W16,
    const float* __restrict__ bias, _Float16* sX, _Float16* dst,
    int b, int tid, int wave, int lane, int g, int ncol) {
  // stage x window (64x256 f32 -> f16 LDS, stride SX_STRIDE)
  const float* xp = x + (size_t)b * (N_TOK * D_MODEL);
  for (int i = tid; i < N_TOK * (D_MODEL / 4); i += 256) {
    int r  = i >> 6;           // token row
    int c4 = (i & 63) * 4;     // channel
    v4f x4 = *(const v4f*)(xp + r * D_MODEL + c4);
    _Float16* d = sX + r * SX_STRIDE + c4;
    d[0] = (_Float16)x4.x; d[1] = (_Float16)x4.y;
    d[2] = (_Float16)x4.z; d[3] = (_Float16)x4.w;
  }
  __syncthreads();

  // 64 output tiles (4 row-blocks x 16 col tiles); wave owns cols {wave, wave+8}
  for (int tm = 0; tm < 4; ++tm) {
    v16h afr[8];                               // hoist A-frags: reused for 2 tiles
    #pragma unroll
    for (int ks = 0; ks < 8; ++ks)
      afr[ks] = load_a_frag(sX, SX_STRIDE, tm * 16, ks * 32, lane);
    #pragma unroll
    for (int half = 0; half < 2; ++half) {
      int tn = wave + half * 8;
      v8f acc = {};
      #pragma unroll
      for (int ks = 0; ks < 8; ++ks)
        acc = wmma_f16(afr[ks], load_b_frag(W16, D_MODEL, tn * 16, ks * 32, lane), acc);
      int col = tn * 16 + ncol;
      float bb = bias[col];
      #pragma unroll
      for (int vv = 0; vv < 8; ++vv) {
        int m = tm * 16 + vv + 8 * g;
        float val = acc[vv] + bb;
        if (MODE == 0)      dst[m * SX_STRIDE + col]   = (_Float16)(val * ATTN_SCALE);
        else if (MODE == 1) dst[m * SX_STRIDE + col]   = (_Float16)val;
        else                dst[col * SVT_STRIDE + m]  = (_Float16)val;  // transposed
      }
    }
  }
  __syncthreads();
}

// ---------------------------------------------------------------------------
// Main kernel: one block (8 wave32) per window.
// ---------------------------------------------------------------------------
__global__ __launch_bounds__(256)
void wmha_kernel(const float* __restrict__ q, const float* __restrict__ k,
                 const float* __restrict__ v, const float* __restrict__ mask,
                 const _Float16* __restrict__ Wq16, const _Float16* __restrict__ Wk16,
                 const _Float16* __restrict__ Wv16, const _Float16* __restrict__ Wp16,
                 const float* __restrict__ bq, const float* __restrict__ bk,
                 const float* __restrict__ bv, const float* __restrict__ bp,
                 float* __restrict__ out_feats, float* __restrict__ out_attn) {
  __shared__ __align__(16) unsigned char smem[SMEM_BYTES];
  _Float16* sX  = (_Float16*)smem;              // proj input staging (region0)
  float*    sS  = (float*)smem;                 // scores f32        (region0)
  _Float16* sF2 = (_Float16*)smem;              // shuffled feats    (region0)
  _Float16* sQ  = (_Float16*)(smem + OFF_SQ);   // qh (pre-scaled) f16
  _Float16* sK  = (_Float16*)(smem + OFF_SK);   // kh f16
  _Float16* sVT = (_Float16*)(smem + OFF_SVT);  // vh transposed f16 (256x64)
  _Float16* sA  = (_Float16*)(smem + OFF_SA);   // attn f16 (64x64)

  const int b    = blockIdx.x;
  const int tid  = threadIdx.x;
  const int wave = tid >> 5;
  const int lane = tid & 31;
  const int g    = lane >> 4;
  const int ncol = lane & 15;

  // ========== Phase 1: projections qh / kh / vh (specialized passes) ======
  projection_pass<0>(q, Wq16, bq, sX, sQ,  b, tid, wave, lane, g, ncol);
  projection_pass<1>(k, Wk16, bk, sX, sK,  b, tid, wave, lane, g, ncol);
  projection_pass<2>(v, Wv16, bv, sX, sVT, b, tid, wave, lane, g, ncol);

  // ========== Phase 2: scores S = qh . kh^T (64x64, K=256) ================
  // wave -> col tile tn = wave&3, row tiles {wave>>2, (wave>>2)+2}; B-frags hoisted.
  {
    int tn  = wave & 3;
    int tm0 = wave >> 2;
    v16h bfr[8];
    #pragma unroll
    for (int ks = 0; ks < 8; ++ks)
      bfr[ks] = load_b_frag(sK, SX_STRIDE, tn * 16, ks * 32, lane); // B[c][j]=kh[j][c]
    #pragma unroll
    for (int half = 0; half < 2; ++half) {
      int tm = tm0 + half * 2;
      v8f acc = {};
      #pragma unroll
      for (int ks = 0; ks < 8; ++ks)
        acc = wmma_f16(load_a_frag(sQ, SX_STRIDE, tm * 16, ks * 32, lane), bfr[ks], acc);
      #pragma unroll
      for (int vv = 0; vv < 8; ++vv) {
        int m = tm * 16 + vv + 8 * g;
        sS[m * SS_STRIDE + tn * 16 + ncol] = acc[vv];
      }
    }
  }
  __syncthreads();

  // ========== Phase 3: mask + softmax + mask (fp32, one thread per row) ===
  if (tid < N_TOK) {
    const float* mrow = mask + tid * N_TOK;
    float* srow = sS + tid * SS_STRIDE;
    float mx = -3.0e38f;
    for (int j = 0; j < N_TOK; ++j) {
      float s = srow[j] - 1.0e6f * mrow[j];   // additive pre-softmax mask
      srow[j] = s;
      mx = fmaxf(mx, s);
    }
    float sum = 0.0f;
    for (int j = 0; j < N_TOK; ++j) {
      float e = __expf(srow[j] - mx);
      srow[j] = e;
      sum += e;
    }
    float inv = 1.0f / sum;
    for (int j = 0; j < N_TOK; ++j) {
      float post = srow[j] * inv * mrow[j];   // multiplicative post-softmax mask
      srow[j] = post;
      sA[tid * SA_STRIDE + j] = (_Float16)post;
    }
  }
  __syncthreads();

  // cooperative, coalesced writeout of attn (output #2)
  {
    float* oa = out_attn + (size_t)b * (N_TOK * N_TOK);
    for (int i = tid; i < N_TOK * N_TOK; i += 256) {
      oa[i] = sS[(i >> 6) * SS_STRIDE + (i & 63)];
    }
  }
  __syncthreads();

  // ========== Phase 4: feats = attn . vh (64x256, K=64), store shuffled ===
  // torch transpose(1,2).reshape bug: feats2.flat[n*64+m] = feats[m][n]
  for (int tm = 0; tm < 4; ++tm) {
    v16h afr[2];
    #pragma unroll
    for (int ks = 0; ks < 2; ++ks)
      afr[ks] = load_a_frag(sA, SA_STRIDE, tm * 16, ks * 32, lane);
    #pragma unroll
    for (int half = 0; half < 2; ++half) {
      int tn = wave + half * 8;
      v8f acc = {};
      #pragma unroll
      for (int ks = 0; ks < 2; ++ks)
        acc = wmma_f16(afr[ks], load_b_frag(sVT, SVT_STRIDE, tn * 16, ks * 32, lane), acc);
      int col = tn * 16 + ncol;       // channel n_
      #pragma unroll
      for (int vv = 0; vv < 8; ++vv) {
        int m = tm * 16 + vv + 8 * g; // token m_
        int flat = col * N_TOK + m;   // position in feats^T flat
        sF2[(flat >> 8) * SX_STRIDE + (flat & 255)] = (_Float16)acc[vv];
      }
    }
  }
  __syncthreads();

  // ========== Phase 5: out = feats2 . Wp^T + bp (64x256, K=256) ===========
  float* of = out_feats + (size_t)b * (N_TOK * D_MODEL);
  for (int tm = 0; tm < 4; ++tm) {
    v16h afr[8];
    #pragma unroll
    for (int ks = 0; ks < 8; ++ks)
      afr[ks] = load_a_frag(sF2, SX_STRIDE, tm * 16, ks * 32, lane);
    #pragma unroll
    for (int half = 0; half < 2; ++half) {
      int tn = wave + half * 8;
      v8f acc = {};
      #pragma unroll
      for (int ks = 0; ks < 8; ++ks)
        acc = wmma_f16(afr[ks], load_b_frag(Wp16, D_MODEL, tn * 16, ks * 32, lane), acc);
      int col = tn * 16 + ncol;
      float bb = bp[col];
      #pragma unroll
      for (int vv = 0; vv < 8; ++vv) {
        int m = tm * 16 + vv + 8 * g;
        of[m * D_MODEL + col] = acc[vv] + bb;
      }
    }
  }
}

// ---------------------------------------------------------------------------
// Launch
// ---------------------------------------------------------------------------
extern "C" void kernel_launch(void* const* d_in, const int* in_sizes, int n_in,
                              void* d_out, int out_size, void* d_ws, size_t ws_size,
                              hipStream_t stream) {
  (void)in_sizes; (void)n_in; (void)out_size; (void)ws_size;

  // setup_inputs() dict order: v, k, q, mask, Wq, bq, Wk, bk, Wv, bv, Wp, bp
  const float* v    = (const float*)d_in[0];
  const float* k    = (const float*)d_in[1];
  const float* q    = (const float*)d_in[2];
  const float* mask = (const float*)d_in[3];
  const float* Wq   = (const float*)d_in[4];
  const float* bq   = (const float*)d_in[5];
  const float* Wk   = (const float*)d_in[6];
  const float* bk   = (const float*)d_in[7];
  const float* Wv   = (const float*)d_in[8];
  const float* bv   = (const float*)d_in[9];
  const float* Wp   = (const float*)d_in[10];
  const float* bp   = (const float*)d_in[11];

  // d_ws: four 256x256 f16 weight copies (512 KB)
  _Float16* ws   = (_Float16*)d_ws;
  _Float16* Wq16 = ws;
  _Float16* Wk16 = ws + 65536;
  _Float16* Wv16 = ws + 131072;
  _Float16* Wp16 = ws + 196608;

  convert_weights_kernel<<<256, 256, 0, stream>>>(Wq, Wk, Wv, Wp,
                                                  Wq16, Wk16, Wv16, Wp16);

  float* out_feats = (float*)d_out;                                  // 4096*64*256
  float* out_attn  = out_feats + (size_t)N_WIN * N_TOK * D_MODEL;    // 4096*64*64

  wmha_kernel<<<N_WIN, 256, 0, stream>>>(q, k, v, mask,
                                         Wq16, Wk16, Wv16, Wp16,
                                         bq, bk, bv, bp,
                                         out_feats, out_attn);
}
